// Attention_87436944212609
// MI455X (gfx1250) — compile-verified, matrix operand order checked by reading, not run on hardware
//
#include <hip/hip_runtime.h>
#include <hip/hip_bf16.h>

// ---------------------------------------------------------------------------
// BoTNet attention for MI455X (gfx1250, wave32, WMMA bf16 16x16x32).
//   b=4, heads=4, hw=4096, dim_head=64, C=256
// Phase 1: qkv projection GEMM (+fused scale / positional-emb epilogue) -> bf16 ws
// Phase 2: flash attention; K' tiles staged via GLOBAL_LOAD_ASYNC_TO_LDS_B128
//          (ASYNCcnt path), f32 online softmax, bf16 WMMA GEMMs.
// ---------------------------------------------------------------------------

typedef __attribute__((ext_vector_type(16))) __bf16 v16bf;
typedef __attribute__((ext_vector_type(8)))  float  v8f;

#define HW        4096
#define DH        64
#define SCALE_Q   0.125f

// ---- helpers ---------------------------------------------------------------

__device__ __forceinline__ unsigned short f2bf(float f) {
    unsigned u = __builtin_bit_cast(unsigned, f);
    unsigned rnd = 0x7fffu + ((u >> 16) & 1u);   // round-to-nearest-even
    return (unsigned short)((u + rnd) >> 16);
}

__device__ __forceinline__ unsigned pack2bf(float a, float b) {
    return (unsigned)f2bf(a) | ((unsigned)f2bf(b) << 16);
}

union Op16 { v16bf v; unsigned u[8]; };

__device__ __forceinline__ v8f wmma_bf16(v16bf a, v16bf b, v8f c) {
    // (neg_a, A, neg_b, B, c_mod, C, reuse_a, reuse_b)
    return __builtin_amdgcn_wmma_f32_16x16x32_bf16(false, a, false, b,
                                                   (short)0, c, false, false);
}

// reductions within each 16-lane half of a wave32 (matches C-matrix halves)
__device__ __forceinline__ float hmax16(float v) {
    v = fmaxf(v, __shfl_xor(v, 1, 32));
    v = fmaxf(v, __shfl_xor(v, 2, 32));
    v = fmaxf(v, __shfl_xor(v, 4, 32));
    v = fmaxf(v, __shfl_xor(v, 8, 32));
    return v;
}
__device__ __forceinline__ float hsum16(float v) {
    v += __shfl_xor(v, 1, 32);
    v += __shfl_xor(v, 2, 32);
    v += __shfl_xor(v, 4, 32);
    v += __shfl_xor(v, 8, 32);
    return v;
}

// A-operand k-pair index for 16-bit 16x32 A layout (ISA 7.12.2):
// lane: m = lane&15, hb = lane>>4 ; VGPR v holds bf16 pair (k, k+1)
__device__ __forceinline__ int kpair(int v, int hb) {
    return ((v >= 4) ? 16 : 0) + hb * 8 + (v & 3) * 2;
}

// async global -> LDS copy of 16 bytes (GV mode, tracked on ASYNCcnt)
__device__ __forceinline__ void async_ld_b128(const void* gaddr, void* lds) {
    unsigned loff = (unsigned)(size_t)lds;               // low 32 bits = LDS addr
    unsigned long long ga = (unsigned long long)(size_t)gaddr;
    asm volatile("global_load_async_to_lds_b128 %0, %1, off"
                 :: "v"(loff), "v"(ga) : "memory");
}
__device__ __forceinline__ void wait_asynccnt0() {
    asm volatile("s_wait_asynccnt 0" ::: "memory");
}

// ---------------------------------------------------------------------------
// Kernel 1: qkv = w_qkv @ fmap  (per batch: M=768, N=4096, K=256)
//   epilogue: q *= 0.125 ; k += pos_height[oc%64, x] + pos_width[y, x]
//   stores bf16 into ws slabs (contiguous (b, chan, s) layout == per-head
//   4096x64 row-major matrices thanks to the raw-reshape identity).
// ---------------------------------------------------------------------------
#define PBM 64
#define PBN 64
#define PBK 32
#define ASTR 36   // bf16 elems per LDS row (72B, conflict-free & 4B aligned)
#define BSTR 36

__global__ __launch_bounds__(256)
void qkv_proj_kernel(const float* __restrict__ fmap,
                     const float* __restrict__ wq,
                     const float* __restrict__ ph,
                     const float* __restrict__ pwd,
                     unsigned short* __restrict__ qws,
                     unsigned short* __restrict__ kws,
                     unsigned short* __restrict__ vws)
{
    __shared__ unsigned short a_lds[PBM * ASTR];   // [m][k]  (K-contiguous)
    __shared__ unsigned short b_lds[PBN * BSTR];   // [n][k]  (K-contiguous)

    const int b   = blockIdx.z;
    const int o0  = blockIdx.y * PBM;      // output-channel base, 0..704
    const int s0  = blockIdx.x * PBN;      // spatial base
    const int tid = threadIdx.x;
    const int wave = tid >> 5, lane = tid & 31;
    const int l16 = lane & 15, hb = lane >> 4;

    const int mi = (wave & 3) * 16;        // this wave's M-tile
    const int nj = (wave >> 2) * 32;       // this wave's two N-tiles

    v8f acc[2] = {};

    for (int kb = 0; kb < 256; kb += PBK) {
        __syncthreads();
        {   // A tile: w_qkv rows [o0, o0+64), cols [kb, kb+32)  -> bf16 pairs
            int mm = tid >> 2;
            int kk = (tid & 3) * 8;
            const float* src = wq + (o0 + mm) * 256 + kb + kk;
            unsigned* dst = (unsigned*)&a_lds[mm * ASTR + kk];
#pragma unroll
            for (int p = 0; p < 4; ++p)
                dst[p] = pack2bf(src[2 * p], src[2 * p + 1]);
        }
        {   // B tile: fmap[b][kb+kk][s0+nn] stored transposed [n][k]
            int kk  = tid >> 3;
            int nn0 = (tid & 7) * 8;
            const float* src = fmap + ((size_t)b * 256 + kb + kk) * (size_t)HW + s0 + nn0;
#pragma unroll
            for (int p = 0; p < 8; ++p)
                b_lds[(nn0 + p) * BSTR + kk] = f2bf(src[p]);
        }
        __syncthreads();

        Op16 aa;
#pragma unroll
        for (int v = 0; v < 8; ++v)
            aa.u[v] = *(const unsigned*)&a_lds[(mi + l16) * ASTR + kpair(v, hb)];
#pragma unroll
        for (int t = 0; t < 2; ++t) {
            Op16 bb;
            int n = nj + t * 16 + l16;
#pragma unroll
            for (int v = 0; v < 8; ++v)
                bb.u[v] = *(const unsigned*)&b_lds[n * BSTR + kpair(v, hb)];
            acc[t] = wmma_bf16(aa.v, bb.v, acc[t]);
        }
    }

    // Epilogue: C element (m,n): lane = n + 16*(m>=8), vgpr r = m&7
    const int region = o0 >> 8;            // 0=q, 1=k, 2=v (BM=64 never straddles)
#pragma unroll
    for (int t = 0; t < 2; ++t) {
#pragma unroll
        for (int r = 0; r < 8; ++r) {
            int m = r + hb * 8;
            int o = o0 + mi + m;
            int s = s0 + nj + t * 16 + l16;
            int x = s & 63, y = s >> 6;
            float val = acc[t][r];
            unsigned short* dst;
            if (region == 0) {
                val *= SCALE_Q;
                dst = qws + ((size_t)b * 256 + o) * (size_t)HW + s;
            } else if (region == 1) {
                int ok = o - 256;
                val += ph[(ok & 63) * 64 + x] + pwd[y * 64 + x];
                dst = kws + ((size_t)b * 256 + ok) * (size_t)HW + s;
            } else {
                dst = vws + ((size_t)b * 256 + (o - 512)) * (size_t)HW + s;
            }
            *dst = f2bf(val);
        }
    }
}

// ---------------------------------------------------------------------------
// Kernel 2: flash attention over one (b,h) slab (4096x64), 128 rows per block,
// 8 waves x 16 rows. Online softmax in f32, GEMMs via bf16 WMMA.
// K' tile: async global->LDS (ASYNCcnt). V tile: VGPR path (needs transpose).
// ---------------------------------------------------------------------------
#define BM    128
#define BN    64
#define KSTRK 72   // K' tile stride: 144B rows (16B aligned for b128 async st,
                   //  bank step 36 -> 16 distinct banks for j=0..15)
#define KSTR  68   // V / P strides: 136B rows (4B aligned, conflict-free)

__global__ __launch_bounds__(256)
void fa_kernel(const unsigned short* __restrict__ qws,
               const unsigned short* __restrict__ kws,
               const unsigned short* __restrict__ vws,
               float* __restrict__ out)
{
    __shared__ unsigned short k_lds[BN * KSTRK];        // K' tile, row-major [j][d]
    __shared__ unsigned short v_lds[DH * KSTR];         // V tile, transposed [d][j]
    __shared__ unsigned short p_lds[8 * 16 * KSTR];     // per-wave P scratch

    const int slab = blockIdx.y;                        // b*4 + head
    const int row0 = blockIdx.x * BM;
    const int tid  = threadIdx.x;
    const int wave = tid >> 5, lane = tid & 31;
    const int l16  = lane & 15, hb = lane >> 4;

    const size_t slabOff = (size_t)slab * (HW * DH);
    const unsigned short* Qs = qws + slabOff;
    const unsigned short* Ks = kws + slabOff;
    const unsigned short* Vs = vws + slabOff;

    // Q for this wave's 16 rows, A-layout registers, two 32-wide k-steps over d
    const int qrow = row0 + wave * 16 + l16;
    Op16 qa[2];
#pragma unroll
    for (int ks = 0; ks < 2; ++ks)
#pragma unroll
        for (int v = 0; v < 8; ++v) {
            int d = ks * 32 + kpair(v, hb);
            qa[ks].u[v] = *(const unsigned*)(Qs + (size_t)qrow * DH + d);
        }

    v8f oacc[4] = {};
    float mrow[8], lrow[8];
#pragma unroll
    for (int r = 0; r < 8; ++r) { mrow[r] = -3.0e38f; lrow[r] = 0.f; }

    unsigned short* pw = &p_lds[wave * 16 * KSTR];

    for (int jt = 0; jt < HW / BN; ++jt) {
        __syncthreads();
        {   // ---- K' tile: async global->LDS, 2x b128 per thread ----
            int jj = tid >> 2;              // 0..63
            int d0 = (tid & 3) * 16;        // 0,16,32,48
            const unsigned short* kg = Ks + ((size_t)(jt * BN + jj)) * DH + d0;
            async_ld_b128(kg,     &k_lds[jj * KSTRK + d0]);
            async_ld_b128(kg + 8, &k_lds[jj * KSTRK + d0 + 8]);

            // ---- V tile: load + transpose into LDS ----
            const unsigned* vsrc = (const unsigned*)(Vs + ((size_t)(jt * BN + jj)) * DH + d0);
#pragma unroll
            for (int p = 0; p < 8; ++p) {
                unsigned pv = vsrc[p];
                int d = d0 + p * 2;
                v_lds[d * KSTR + jj]       = (unsigned short)(pv & 0xffffu);
                v_lds[(d + 1) * KSTR + jj] = (unsigned short)(pv >> 16);
            }
            if (jt + 1 < HW / BN)   // speculative prefetch of next V tile
                __builtin_prefetch(Vs + ((size_t)((jt + 1) * BN + jj)) * DH + d0, 0, 1);
        }
        wait_asynccnt0();
        __syncthreads();

        // ---- S = Q x K'^T  (16 x 64 per wave) ----
        v8f sacc[4];
#pragma unroll
        for (int nt = 0; nt < 4; ++nt) {
            v8f c = {};
#pragma unroll
            for (int ks = 0; ks < 2; ++ks) {
                Op16 bb;
                int j = nt * 16 + l16;
#pragma unroll
                for (int v = 0; v < 8; ++v)
                    bb.u[v] = *(const unsigned*)&k_lds[j * KSTRK + ks * 32 + kpair(v, hb)];
                c = wmma_bf16(qa[ks].v, bb.v, c);
            }
            sacc[nt] = c;
        }

        // ---- online softmax (rows live in 16-lane halves) ----
        float corr[8];
#pragma unroll
        for (int r = 0; r < 8; ++r) {
            float mx = fmaxf(fmaxf(sacc[0][r], sacc[1][r]),
                             fmaxf(sacc[2][r], sacc[3][r]));
            mx = hmax16(mx);
            float mnew = fmaxf(mrow[r], mx);
            corr[r] = __expf(mrow[r] - mnew);
            mrow[r] = mnew;
            float rs = 0.f;
#pragma unroll
            for (int nt = 0; nt < 4; ++nt) {
                float p = __expf(sacc[nt][r] - mnew);
                sacc[nt][r] = p;
                rs += p;
            }
            lrow[r] = lrow[r] * corr[r] + hsum16(rs);
        }
#pragma unroll
        for (int dt = 0; dt < 4; ++dt)
#pragma unroll
            for (int r = 0; r < 8; ++r)
                oacc[dt][r] = oacc[dt][r] * corr[r];

        // ---- P: C-layout -> LDS -> A-layout (per-wave private scratch) ----
#pragma unroll
        for (int nt = 0; nt < 4; ++nt)
#pragma unroll
            for (int r = 0; r < 8; ++r)
                pw[(r + hb * 8) * KSTR + nt * 16 + l16] = f2bf(sacc[nt][r]);
        asm volatile("s_wait_dscnt 0" ::: "memory");

        // ---- O += P x V ----
#pragma unroll
        for (int ks = 0; ks < 2; ++ks) {
            Op16 pa;
#pragma unroll
            for (int v = 0; v < 8; ++v)
                pa.u[v] = *(const unsigned*)&pw[l16 * KSTR + ks * 32 + kpair(v, hb)];
#pragma unroll
            for (int dt = 0; dt < 4; ++dt) {
                Op16 vb;
                int d = dt * 16 + l16;
#pragma unroll
                for (int v = 0; v < 8; ++v)
                    vb.u[v] = *(const unsigned*)&v_lds[d * KSTR + ks * 32 + kpair(v, hb)];
                oacc[dt] = wmma_bf16(pa.v, vb.v, oacc[dt]);
            }
        }
    }

    // ---- finalize: O /= l, store f32 (output reshape is the identity slab map)
    float* outSlab = out + slabOff;
#pragma unroll
    for (int r = 0; r < 8; ++r) {
        float inv = 1.0f / lrow[r];
        int i = row0 + wave * 16 + r + hb * 8;
#pragma unroll
        for (int dt = 0; dt < 4; ++dt)
            outSlab[(size_t)i * DH + dt * 16 + l16] = oacc[dt][r] * inv;
    }
}

// ---------------------------------------------------------------------------
extern "C" void kernel_launch(void* const* d_in, const int* in_sizes, int n_in,
                              void* d_out, int out_size, void* d_ws, size_t ws_size,
                              hipStream_t stream)
{
    (void)in_sizes; (void)n_in; (void)out_size; (void)ws_size;
    const float* fmap = (const float*)d_in[0];   // (4,256,64,64)
    const float* wqkv = (const float*)d_in[1];   // (768,256)
    const float* ph   = (const float*)d_in[2];   // (64,64)
    const float* pwd  = (const float*)d_in[3];   // (64,64)
    float* out = (float*)d_out;

    unsigned short* ws  = (unsigned short*)d_ws;
    unsigned short* qws = ws;                         // 4*256*4096 bf16 = 8 MB
    unsigned short* kws = ws + (size_t)4 * 256 * HW;  // k + emb
    unsigned short* vws = ws + (size_t)8 * 256 * HW;  // 24 MB total

    dim3 g1(HW / PBN, 768 / PBM, 4);
    qkv_proj_kernel<<<g1, 256, 0, stream>>>(fmap, wqkv, ph, pwd, qws, kws, vws);

    dim3 g2(HW / BM, 16);
    fa_kernel<<<g2, 256, 0, stream>>>(qws, kws, vws, out);
}
